// DGCNN_exit1_84911503442330
// MI455X (gfx1250) — matrix operand order repaired and checked by reading, not compiled, not run on hardware
//
#include <hip/hip_runtime.h>
#include <hip/hip_bf16.h>
#include <math.h>

// ---------------------------------------------------------------------------
// DGCNN head for MI455X (gfx1250, wave32).
//  - topk fused with pairwise distances (avoids 134MB pd tensor)
//  - exit conv: LDS-staged WMMA, h1 slab fetched via global_load_async_to_lds
//  - fc2/p1..p4: generic WMMA fc kernel (bias+BN+leaky fused epilogue)
// ---------------------------------------------------------------------------

typedef __attribute__((ext_vector_type(16))) _Float16 v16h;
typedef __attribute__((ext_vector_type(8)))  float    v8f;

#define BB   32
#define NN   1024
#define KNN  20

__device__ __forceinline__ float leaky_(float x){ return x >= 0.f ? x : 0.2f*x; }

__device__ __forceinline__ unsigned hash_u32(unsigned x){
  x ^= x >> 16; x *= 0x7feb352dU; x ^= x >> 15; x *= 0x846ca68bU; x ^= x >> 16; return x;
}
// Deterministic N(0,1) stand-in for jax.random.normal(key=42) (compile-only loop,
// threefry not reproduced bit-exact).
__device__ __forceinline__ float gauss_hash(unsigned key){
  unsigned h1 = hash_u32(key ^ 0x9e3779b9u);
  unsigned h2 = hash_u32(h1 + 0x85ebca6bu);
  float u1 = (float)(h1 >> 8) * (1.0f/16777216.0f) + 1e-7f;
  float u2 = (float)(h2 >> 8) * (1.0f/16777216.0f);
  return sqrtf(-2.0f*__logf(u1)) * __cosf(6.28318530718f*u2);
}

// WMMA fragment K-index map for 16-bit A fragments (ISA 7.12.2, wave32):
// lanes 0-15 hold K={0..7,16..23}, lanes 16-31 hold K={8..15,24..31}.
__device__ __forceinline__ int kmapA(int e, int hi) {
  return (e < 8) ? (hi*8 + e) : (16 + hi*8 + (e - 8));
}

// ---------------------------------------------------------------------------
// K1: fused pairwise distance + top-20 indices. 4 waves/block, 1 row per wave.
// ---------------------------------------------------------------------------
__global__ __launch_bounds__(128)
void topk_kernel(const float* __restrict__ x, int* __restrict__ idxout)
{
  __shared__ float xs[3*NN];
  __shared__ float xx[NN];
  __shared__ float cand[4*NN];
  const int b     = blockIdx.x >> 8;        // 256 blocks per batch
  const int nbase = (blockIdx.x & 255) << 2;
  const float* xb = x + (size_t)b*3*NN;
  for (int i = threadIdx.x; i < NN; i += 128) {
    float v0 = xb[i], v1 = xb[NN+i], v2 = xb[2*NN+i];
    xs[i]=v0; xs[NN+i]=v1; xs[2*NN+i]=v2;
    xx[i] = v0*v0 + v1*v1 + v2*v2;
  }
  __syncthreads();
  const int wave = threadIdx.x >> 5, lane = threadIdx.x & 31;
  const int n = nbase + wave;
  const float a0=xs[n], a1=xs[NN+n], a2=xs[2*NN+n], xn=xx[n];
  float* cv = cand + wave*NN;
  for (int m = lane; m < NN; m += 32)
    cv[m] = 2.f*(a0*xs[m] + a1*xs[NN+m] + a2*xs[2*NN+m]) - xn - xx[m];
  __syncthreads();
  int* orow = idxout + (size_t)(b*NN + n)*KNN;
  for (int r = 0; r < KNN; ++r) {
    float best = -3.0e38f; int bi = 0;
    for (int m = lane; m < NN; m += 32) {
      float d = cv[m];
      if (d > best) { best = d; bi = m; }
    }
    for (int off = 16; off > 0; off >>= 1) {
      float ov = __shfl_down(best, off, 32);
      int   oi = __shfl_down(bi,   off, 32);
      if (ov > best) { best = ov; bi = oi; }
    }
    if (lane == 0) { orow[r] = bi; cv[bi] = -3.0e38f; }
    __syncthreads();
  }
}

// ---------------------------------------------------------------------------
// K2: edge conv 6->64 + BN + leaky + max over 20 neighbors. One wave per point,
// each lane owns 2 output channels. Weights + folded BN in LDS.
// ---------------------------------------------------------------------------
__global__ __launch_bounds__(256)
void edgeconv_kernel(const float* __restrict__ x, const int* __restrict__ idx,
                     const float* __restrict__ w,
                     const float* __restrict__ g, const float* __restrict__ bta,
                     const float* __restrict__ mu, const float* __restrict__ var,
                     float* __restrict__ h1)
{
  __shared__ float xs[3*NN];
  __shared__ float wsm[64*6];
  __shared__ float sc[64], sh[64];
  const int b     = blockIdx.x >> 7;        // 128 blocks per batch
  const int nbase = (blockIdx.x & 127) << 3;
  const float* xb = x + (size_t)b*3*NN;
  for (int i = threadIdx.x; i < NN; i += 256) {
    xs[i]=xb[i]; xs[NN+i]=xb[NN+i]; xs[2*NN+i]=xb[2*NN+i];
  }
  for (int i = threadIdx.x; i < 64*6; i += 256) wsm[i] = w[i];
  if (threadIdx.x < 64) {
    float s = g[threadIdx.x] * rsqrtf(var[threadIdx.x] + 1e-5f);
    sc[threadIdx.x] = s;
    sh[threadIdx.x] = bta[threadIdx.x] - mu[threadIdx.x]*s;
  }
  __syncthreads();
  const int wave = threadIdx.x >> 5, lane = threadIdx.x & 31;
  const int n = nbase + wave;
  const float c0=xs[n], c1=xs[NN+n], c2=xs[2*NN+n];
  const int* irow = idx + (size_t)(b*NN + n)*KNN;
  const int o0 = lane, o1 = lane + 32;
  const float* w0 = wsm + o0*6;
  const float* w1 = wsm + o1*6;
  float m0 = -3.0e38f, m1 = -3.0e38f;
  for (int k = 0; k < KNN; ++k) {
    int mi = irow[k];
    float e0 = xs[mi]-c0, e1 = xs[NN+mi]-c1, e2 = xs[2*NN+mi]-c2;
    float acc0 = w0[0]*e0 + w0[1]*e1 + w0[2]*e2 + w0[3]*c0 + w0[4]*c1 + w0[5]*c2;
    float acc1 = w1[0]*e0 + w1[1]*e1 + w1[2]*e2 + w1[3]*c0 + w1[4]*c1 + w1[5]*c2;
    m0 = fmaxf(m0, leaky_(sc[o0]*acc0 + sh[o0]));
    m1 = fmaxf(m1, leaky_(sc[o1]*acc1 + sh[o1]));
  }
  h1[((size_t)b*64 + o0)*NN + n] = m0;
  h1[((size_t)b*64 + o1)*NN + n] = m1;
}

// ---------------------------------------------------------------------------
// K3: exit conv via WMMA, LDS-staged.
// Block = 256 threads (8 waves) handles (batch b, 64-wide n-slab):
//   - h1 slab (64 k x 64 n, f32, 16KB) fetched with global_load_async_to_lds_b128
//   - W (256x64) staged once as f16 (32KB)
//   - wave w: n-tile = w&3, o-half = w>>2; B-fragments built once from LDS and
//     reused across 8 unrolled o-tiles -> 16 v_wmma_f32_16x16x32_f16 per wave.
// Single manually-partitioned LDS block => async-copy byte offsets start at 0.
// ---------------------------------------------------------------------------
#define K3_SMEM_H   (64*64*4)            // 16384 B : h1 slab f32 at offset 0
#define K3_SMEM_W   (256*64*2)           // 32768 B : W f16
#define K3_SMEM_SC  (256*4)
__global__ __launch_bounds__(256)
void exitconv_wmma_kernel(const float* __restrict__ h1, const float* __restrict__ w,
                          const float* __restrict__ g, const float* __restrict__ bta,
                          const float* __restrict__ mu, const float* __restrict__ var,
                          float* __restrict__ h2)
{
  __shared__ __align__(16) unsigned char smem[K3_SMEM_H + K3_SMEM_W + 2*K3_SMEM_SC];
  float*    hs  = (float*)smem;                               // LDS offset 0
  _Float16* wsm = (_Float16*)(smem + K3_SMEM_H);
  float*    sc  = (float*)(smem + K3_SMEM_H + K3_SMEM_W);
  float*    sh  = sc + 256;

  const int b  = blockIdx.x >> 4;        // 16 n-slabs per batch
  const int n0 = (blockIdx.x & 15) << 6;

  // ---- async copy of the h1 slab into LDS (4 x B128 per thread) ----
  const float* src = h1 + (size_t)b*64*NN + n0;
  #pragma unroll
  for (int j = 0; j < 4; ++j) {
    int q = threadIdx.x + j*256;             // float4 id, 0..1023
    int k = q >> 4, c4 = q & 15;             // row k, 16-byte chunk in row
    const float* ga = src + (size_t)k*NN + c4*4;
    unsigned ldsoff = (unsigned)q * 16u;     // byte offset within hs
    asm volatile("global_load_async_to_lds_b128 %0, %1, off"
                 :: "v"(ldsoff), "v"(ga) : "memory");
  }
  // ---- stage W as f16 + folded BN while the DMA is in flight ----
  for (int i = threadIdx.x; i < 256*64; i += 256) wsm[i] = (_Float16)w[i];
  {
    float s = g[threadIdx.x] * rsqrtf(var[threadIdx.x] + 1e-5f);
    sc[threadIdx.x] = s;
    sh[threadIdx.x] = bta[threadIdx.x] - mu[threadIdx.x]*s;
  }
  asm volatile("s_wait_asynccnt 0x0" ::: "memory");
  __syncthreads();

  const int wave = threadIdx.x >> 5, lane = threadIdx.x & 31;
  const int lm = lane & 15, hi = lane >> 4;
  const int nt    = wave & 3;              // n-tile within slab
  const int ohalf = wave >> 2;             // o-tiles 0..7 or 8..15
  const int n  = n0 + nt*16 + lm;

  // B fragments once per wave (f32 LDS -> f16), reused for all 8 o-tiles
  const int kb = hi*16;
  v16h b0, b1;
  #pragma unroll
  for (int e = 0; e < 16; ++e) {
    b0[e] = (_Float16)hs[(kb + e)*64      + nt*16 + lm];
    b1[e] = (_Float16)hs[(32 + kb + e)*64 + nt*16 + lm];
  }

  #pragma unroll
  for (int j = 0; j < 8; ++j) {
    const int ot = ohalf*8 + j;
    const int o  = ot*16 + lm;
    const _Float16* Wr = wsm + o*64;
    v16h a0, a1;
    #pragma unroll
    for (int e = 0; e < 16; ++e) {
      int kk = hi*16 + e;
      a0[e] = Wr[kk];
      a1[e] = Wr[32 + kk];
    }
    v8f c = {};
    c = __builtin_amdgcn_wmma_f32_16x16x32_f16(false, a0, false, b0, (short)0, c, false, false);
    c = __builtin_amdgcn_wmma_f32_16x16x32_f16(false, a1, false, b1, (short)0, c, false, false);
    const int orow = ot*16 + hi*8;
    #pragma unroll
    for (int r = 0; r < 8; ++r) {
      int oo = orow + r;
      h2[((size_t)b*256 + oo)*NN + n] = leaky_(sc[oo]*c[r] + sh[oo]);
    }
  }
}

// ---------------------------------------------------------------------------
// K4: global max+mean pool over n -> feat (B,512) = [max(256) | mean(256)]
// ---------------------------------------------------------------------------
__global__ __launch_bounds__(256)
void pool_kernel(const float* __restrict__ h2, float* __restrict__ feat)
{
  __shared__ float smx[256], ssm[256];
  const int b = blockIdx.x >> 8, o = blockIdx.x & 255;
  const float* p = h2 + ((size_t)b*256 + o)*NN;
  float mx = -3.0e38f, sm = 0.f;
  for (int i = threadIdx.x; i < NN; i += 256) { float v = p[i]; mx = fmaxf(mx, v); sm += v; }
  smx[threadIdx.x] = mx; ssm[threadIdx.x] = sm;
  __syncthreads();
  for (int s = 128; s > 0; s >>= 1) {
    if (threadIdx.x < s) {
      smx[threadIdx.x] = fmaxf(smx[threadIdx.x], smx[threadIdx.x+s]);
      ssm[threadIdx.x] += ssm[threadIdx.x+s];
    }
    __syncthreads();
  }
  if (threadIdx.x == 0) {
    feat[(size_t)b*512 + o]       = smx[0];
    feat[(size_t)b*512 + 256 + o] = ssm[0] * (1.f/NN);
  }
}

// ---------------------------------------------------------------------------
// K5: generic WMMA fc layer: out(32,On) = leaky(bn(in(32,In) @ W^T + bias)).
// One wave per (o-tile, batch-half); K-loop of 16x16x32 f16 WMMAs.
// A = W tile (16 o x 32 i), B = in^T tile (32 i x 16 batch). Bounds-guarded
// so On need not be a multiple of 16 (p4: On=40). In must be %32 (all are).
// ---------------------------------------------------------------------------
__global__ __launch_bounds__(256)
void fc_wmma_kernel(const float* __restrict__ in, const float* __restrict__ w,
                    const float* __restrict__ bias,
                    const float* __restrict__ g, const float* __restrict__ bta,
                    const float* __restrict__ mu, const float* __restrict__ var,
                    float* __restrict__ out, int In, int On, int OnTiles)
{
  const int wave = threadIdx.x >> 5, lane = threadIdx.x & 31;
  const int t = blockIdx.x*8 + wave;
  if (t >= OnTiles*2) return;              // whole-wave exit; EXEC stays full
  const int mt = t & 1;                    // batch half (rows 0-15 / 16-31)
  const int otile = t >> 1;
  const int lm = lane & 15, hi = lane >> 4;

  const int o  = otile*16 + lm;            // A row (may exceed On for p4)
  const int bm = mt*16 + lm;               // B column = batch row
  const bool ovalid = (o < On);
  const float* Wr = w + (size_t)(ovalid ? o : 0)*In;
  const float* Ir = in + (size_t)bm*In;
  const int kb = hi*16;

  v8f c = {};
  const int ksteps = In >> 5;
  for (int ks = 0; ks < ksteps; ++ks) {
    __builtin_prefetch(Wr + (ks+1)*32, 0, 1);   // global_prefetch_b8
    v16h a, bfrag;
    #pragma unroll
    for (int e = 0; e < 16; ++e) {
      int kk = ks*32 + kb + e;
      a[e] = ovalid ? (_Float16)Wr[kk] : (_Float16)0.f;
      bfrag[e] = (_Float16)Ir[ks*32 + kb + e];
    }
    c = __builtin_amdgcn_wmma_f32_16x16x32_f16(false, a, false, bfrag, (short)0, c, false, false);
  }
  const int orow = otile*16 + hi*8;
  #pragma unroll
  for (int r = 0; r < 8; ++r) {
    int oo = orow + r;
    if (oo < On) {
      float s = g[oo]*rsqrtf(var[oo]+1e-5f);
      float val = s*(c[r] + bias[oo]) + (bta[oo] - mu[oo]*s);
      out[(size_t)bm*On + oo] = leaky_(val);
    }
  }
}

// ---------------------------------------------------------------------------
// K6: row L2-normalize + noise for the fc2 output (32 x 1536).
// ---------------------------------------------------------------------------
__global__ __launch_bounds__(256)
void normnoise_kernel(const float* __restrict__ in, float* __restrict__ out)
{
  __shared__ float red[256];
  const int b = blockIdx.x;
  float v[6]; float sq = 0.f;
  #pragma unroll
  for (int j = 0; j < 6; ++j) {
    v[j] = in[(size_t)b*1536 + threadIdx.x + j*256];
    sq += v[j]*v[j];
  }
  red[threadIdx.x] = sq;
  __syncthreads();
  for (int s = 128; s > 0; s >>= 1) {
    if (threadIdx.x < s) red[threadIdx.x] += red[threadIdx.x+s];
    __syncthreads();
  }
  const float inv = 1.f / fmaxf(sqrtf(red[0]), 1e-12f);
  #pragma unroll
  for (int j = 0; j < 6; ++j) {
    int o = threadIdx.x + j*256;
    out[(size_t)b*1536 + o] = v[j]*inv + 0.1f*gauss_hash((unsigned)(b*1536 + o));
  }
}

// ---------------------------------------------------------------------------
// Launch. Input order = flattened setup_inputs() dict (bn tuples -> g,b,m,v).
// ---------------------------------------------------------------------------
extern "C" void kernel_launch(void* const* d_in, const int* in_sizes, int n_in,
                              void* d_out, int out_size, void* d_ws, size_t ws_size,
                              hipStream_t stream) {
  (void)in_sizes; (void)n_in; (void)out_size; (void)ws_size;
  const float* x      = (const float*)d_in[0];
  const float* wconv1 = (const float*)d_in[1];
  const float* bn1g = (const float*)d_in[2],  *bn1b = (const float*)d_in[3],
             * bn1m = (const float*)d_in[4],  *bn1v = (const float*)d_in[5];
  const float* wexit  = (const float*)d_in[6];
  const float* beg  = (const float*)d_in[7],  *beb = (const float*)d_in[8],
             * bem  = (const float*)d_in[9],  *bev = (const float*)d_in[10];
  const float* wfc2 = (const float*)d_in[11], *bfc2 = (const float*)d_in[12];
  const float* f2g  = (const float*)d_in[13], *f2b = (const float*)d_in[14],
             * f2m  = (const float*)d_in[15], *f2v = (const float*)d_in[16];
  const float* wp1 = (const float*)d_in[17], *bp1 = (const float*)d_in[18];
  const float* p1g = (const float*)d_in[19], *p1b = (const float*)d_in[20],
             * p1m = (const float*)d_in[21], *p1v = (const float*)d_in[22];
  const float* wp2 = (const float*)d_in[23], *bp2 = (const float*)d_in[24];
  const float* p2g = (const float*)d_in[25], *p2b = (const float*)d_in[26],
             * p2m = (const float*)d_in[27], *p2v = (const float*)d_in[28];
  const float* wp3 = (const float*)d_in[29], *bp3 = (const float*)d_in[30];
  const float* p3g = (const float*)d_in[31], *p3b = (const float*)d_in[32],
             * p3m = (const float*)d_in[33], *p3v = (const float*)d_in[34];
  const float* wp4 = (const float*)d_in[35], *bp4 = (const float*)d_in[36];
  const float* p4g = (const float*)d_in[37], *p4b = (const float*)d_in[38],
             * p4m = (const float*)d_in[39], *p4v = (const float*)d_in[40];

  char* ws = (char*)d_ws;
  size_t off = 0;
  int*   idx    = (int*)  (ws + off); off += (size_t)BB*NN*KNN*sizeof(int);   // 2.6 MB
  float* h1     = (float*)(ws + off); off += (size_t)BB*64*NN*sizeof(float);  // 8.4 MB
  float* h2     = (float*)(ws + off); off += (size_t)BB*256*NN*sizeof(float); // 33.6 MB
  float* feat   = (float*)(ws + off); off += (size_t)BB*512*sizeof(float);
  float* f1536a = (float*)(ws + off); off += (size_t)BB*1536*sizeof(float);
  float* f1536b = (float*)(ws + off); off += (size_t)BB*1536*sizeof(float);
  float* f512   = (float*)(ws + off); off += (size_t)BB*512*sizeof(float);
  float* f256   = (float*)(ws + off); off += (size_t)BB*256*sizeof(float);
  float* f128   = (float*)(ws + off); off += (size_t)BB*128*sizeof(float);

  topk_kernel<<<dim3(BB*256), dim3(128), 0, stream>>>(x, idx);
  edgeconv_kernel<<<dim3(BB*128), dim3(256), 0, stream>>>(x, idx, wconv1,
      bn1g, bn1b, bn1m, bn1v, h1);
  exitconv_wmma_kernel<<<dim3(BB*16), dim3(256), 0, stream>>>(h1, wexit,
      beg, beb, bem, bev, h2);
  pool_kernel<<<dim3(BB*256), dim3(256), 0, stream>>>(h2, feat);
  // fc2: 512 -> 1536 (96 o-tiles * 2 batch halves = 192 waves = 24 blocks)
  fc_wmma_kernel<<<dim3(24), dim3(256), 0, stream>>>(feat, wfc2, bfc2,
      f2g, f2b, f2m, f2v, f1536a, 512, 1536, 96);
  normnoise_kernel<<<dim3(BB), dim3(256), 0, stream>>>(f1536a, f1536b);
  // p1: 1536 -> 512 (32 tiles -> 64 waves -> 8 blocks)
  fc_wmma_kernel<<<dim3(8), dim3(256), 0, stream>>>(f1536b, wp1, bp1,
      p1g, p1b, p1m, p1v, f512, 1536, 512, 32);
  // p2: 512 -> 256 (16 tiles -> 4 blocks)
  fc_wmma_kernel<<<dim3(4), dim3(256), 0, stream>>>(f512, wp2, bp2,
      p2g, p2b, p2m, p2v, f256, 512, 256, 16);
  // p3: 256 -> 128 (8 tiles -> 2 blocks)
  fc_wmma_kernel<<<dim3(2), dim3(256), 0, stream>>>(f256, wp3, bp3,
      p3g, p3b, p3m, p3v, f128, 256, 128, 8);
  // p4: 128 -> 40 (3 tiles -> 6 waves -> 1 block)
  fc_wmma_kernel<<<dim3(1), dim3(256), 0, stream>>>(f128, wp4, bp4,
      p4g, p4b, p4m, p4v, (float*)d_out, 128, 40, 3);
}
